// GraphAttentionTransformer_75256416961030
// MI455X (gfx1250) — matrix operand order repaired and checked by reading, not compiled, not run on hardware
//
#include <hip/hip_runtime.h>
#include <hip/hip_bf16.h>

typedef __attribute__((ext_vector_type(16))) _Float16 v16h;
typedef __attribute__((ext_vector_type(8)))  _Float16 v8h;
typedef __attribute__((ext_vector_type(8)))  float    v8f;

#define HEADS 8
#define CH    16
#define FDIM  128   // H*C == F_IN == 128 for every layer
#define NEG_SLOPE 0.2f

// ---------- helpers: order-preserving float<->uint encoding for atomicMax ----------
__device__ __forceinline__ unsigned enc_f(float f) {
    int i = __float_as_int(f);
    return (i < 0) ? ~(unsigned)i : ((unsigned)i | 0x80000000u);
}
__device__ __forceinline__ float dec_f(unsigned u) {
    int i = (u & 0x80000000u) ? (int)(u & 0x7FFFFFFFu) : (int)~u;
    return __int_as_float(i);
}

// ---------- f32 -> f16 conversion ----------
__global__ void cvt_f32_f16(const float* __restrict__ in, _Float16* __restrict__ out, int n) {
    int i = blockIdx.x * blockDim.x + threadIdx.x;
    if (i < n) out[i] = (_Float16)in[i];
}

// ---------- weight pre-shuffle into WMMA fragment-major layout ----------
// W: [128, ncols] row-major f32.  out element
//   out[((ct*4+kk)*32+lane)*16+e] = W[kk*32 + (lane>>4)*16 + e, ct*16 + (lane&15)]
// so each lane's B fragment (v16h) is 16 contiguous halves.
__global__ void shuffle_w(const float* __restrict__ W, _Float16* __restrict__ out, int ncols) {
    int i = blockIdx.x * blockDim.x + threadIdx.x;
    int total = (ncols / 16) * 4 * 32 * 16;
    if (i >= total) return;
    const int e    = i & 15;
    const int lane = (i >> 4) & 31;
    const int kk   = (i >> 9) & 3;
    const int ct   = i >> 11;
    const int k    = kk * 32 + (lane >> 4) * 16 + e;
    const int col  = ct * 16 + (lane & 15);
    out[i] = (_Float16)W[k * ncols + col];
}

// ---------- per-layer init: zero aggregation buffer, -inf max, zero denom ----------
__global__ void init_layer(float* __restrict__ outbuf, unsigned* __restrict__ emax,
                           float* __restrict__ denom, int n_nodes) {
    int i = blockIdx.x * blockDim.x + threadIdx.x;
    if (i < n_nodes * FDIM) outbuf[i] = 0.0f;
    if (i < n_nodes * HEADS) {
        emax[i]  = enc_f(-__builtin_inff());
        denom[i] = 0.0f;
    }
}

// ---------- WMMA GEMM: C[nrows,128] = A16[nrows,128] @ W (shuffled) ----------
// 256 threads = 8 waves; wave w owns row tile blockIdx.x*8+w; whole 32KB B staged in LDS.
// A fragments loaded ONCE (4 x v16h in VGPRs), reused across all 8 column tiles -> 32 WMMAs/wave.
__global__ void __launch_bounds__(256) gemm128_wmma(const _Float16* __restrict__ A16,
                                                    const _Float16* __restrict__ Bshuf,
                                                    float* __restrict__ Cout, int nrows) {
    __shared__ _Float16 Bs[16384];                       // 8 col-tiles * 4 k-steps * 512 halves
    {
        ulonglong2*       dst = (ulonglong2*)Bs;
        const ulonglong2* src = (const ulonglong2*)Bshuf;
        for (int t = threadIdx.x; t < 2048; t += 256) dst[t] = src[t];
    }
    __syncthreads();
    const int lane = threadIdx.x & 31;
    const int wave = threadIdx.x >> 5;
    const int row0 = (blockIdx.x * 8 + wave) * 16;
    if (row0 >= nrows) return;                            // wave-uniform: EXEC stays all-1s
    const int m = lane & 15, half = lane >> 4;
    const _Float16* arow = A16 + (size_t)(row0 + m) * FDIM;

    v16h afr[4];                                          // A 16x32 fragments for K=0..127
#pragma unroll
    for (int kk = 0; kk < 4; ++kk) {
        const v8h lo = *(const v8h*)(arow + kk * 32 + half * 16);
        const v8h hi = *(const v8h*)(arow + kk * 32 + half * 16 + 8);
#pragma unroll
        for (int i = 0; i < 8; ++i) { afr[kk][i] = lo[i]; afr[kk][8 + i] = hi[i]; }
    }
#pragma unroll
    for (int ct = 0; ct < 8; ++ct) {
        v8f acc = {};
#pragma unroll
        for (int kk = 0; kk < 4; ++kk) {
            const v16h b = *(const v16h*)(Bs + ((ct * 4 + kk) * 32 + lane) * 16);
            acc = __builtin_amdgcn_wmma_f32_16x16x32_f16(false, afr[kk], false, b, (short)0,
                                                         acc, false, false);
        }
        float* cp = Cout + (size_t)row0 * FDIM + ct * 16 + m;
#pragma unroll
        for (int r = 0; r < 8; ++r) cp[(size_t)(r + half * 8) * FDIM] = acc[r];
    }
}

// ---------- final FC with WMMA: Out[nrows,16] = A16[nrows,128] @ fc_w (shuffled) + b ----------
__global__ void __launch_bounds__(256) fc_wmma(const _Float16* __restrict__ A16,
                                               const _Float16* __restrict__ Bshuf,
                                               const float* __restrict__ bias,
                                               float* __restrict__ Out, int nrows) {
    __shared__ _Float16 Bs[2048];                         // 1 col-tile * 4 k-steps * 512 halves
    {
        ulonglong2*       dst = (ulonglong2*)Bs;
        const ulonglong2* src = (const ulonglong2*)Bshuf;
        for (int t = threadIdx.x; t < 256; t += 256) dst[t] = src[t];
    }
    __syncthreads();
    const int lane = threadIdx.x & 31;
    const int wave = threadIdx.x >> 5;
    const int row0 = (blockIdx.x * 8 + wave) * 16;
    if (row0 >= nrows) return;
    const int m = lane & 15, half = lane >> 4;
    const _Float16* arow = A16 + (size_t)(row0 + m) * FDIM;
    v8f acc = {};
#pragma unroll
    for (int kk = 0; kk < 4; ++kk) {
        const v8h lo = *(const v8h*)(arow + kk * 32 + half * 16);
        const v8h hi = *(const v8h*)(arow + kk * 32 + half * 16 + 8);
        v16h a;
#pragma unroll
        for (int i = 0; i < 8; ++i) { a[i] = lo[i]; a[8 + i] = hi[i]; }
        const v16h b = *(const v16h*)(Bs + (kk * 32 + lane) * 16);
        acc = __builtin_amdgcn_wmma_f32_16x16x32_f16(false, a, false, b, (short)0, acc,
                                                     false, false);
    }
    const float bv = bias[m];
#pragma unroll
    for (int r = 0; r < 8; ++r)
        Out[(size_t)(row0 + r + half * 8) * CH + m] = acc[r] + bv;
}

// ---------- per-node attention scores: e_src/e_dst[n,h] = h[n,h,:] . a_{src,dst}[h,:] ----------
__global__ void escore(const float* __restrict__ h, const float* __restrict__ a_src,
                       const float* __restrict__ a_dst, float* __restrict__ e_src,
                       float* __restrict__ e_dst, int n_nodes) {
    int i = blockIdx.x * blockDim.x + threadIdx.x;   // i = n*8 + head
    if (i >= n_nodes * HEADS) return;
    const int head = i & 7;
    const int n = i >> 3;
    const float* hp = h + (size_t)n * FDIM + head * CH;
    const float* as = a_src + head * CH;
    const float* ad = a_dst + head * CH;
    float s = 0.f, d = 0.f;
#pragma unroll
    for (int c = 0; c < CH; ++c) { const float v = hp[c]; s += v * as[c]; d += v * ad[c]; }
    e_src[i] = s;
    e_dst[i] = d;
}

__device__ __forceinline__ void edge_sd(const long long* __restrict__ ei, long long Eorig,
                                        long long eid, int* s, int* d) {
    if (eid < Eorig) { *s = (int)ei[eid]; *d = (int)ei[Eorig + eid]; }
    else             { const int n = (int)(eid - Eorig); *s = n; *d = n; }  // self-loop
}

__device__ __forceinline__ float leaky(float v) { return v > 0.f ? v : NEG_SLOPE * v; }

// ---------- edge pass 1: segment max ----------
__global__ void edge_max(const long long* __restrict__ ei, long long Eorig, long long Etot,
                         const float* __restrict__ e_src, const float* __restrict__ e_dst,
                         unsigned* __restrict__ emax) {
    long long i = (long long)blockIdx.x * blockDim.x + threadIdx.x;
    if (i >= Etot * HEADS) return;
    const int head = (int)(i & 7);
    int s, d; edge_sd(ei, Eorig, i >> 3, &s, &d);
    const float v = leaky(e_src[s * HEADS + head] + e_dst[d * HEADS + head]);
    atomicMax(&emax[d * HEADS + head], enc_f(v));
}

// ---------- edge pass 2: exp-sum ----------
__global__ void edge_sum(const long long* __restrict__ ei, long long Eorig, long long Etot,
                         const float* __restrict__ e_src, const float* __restrict__ e_dst,
                         const unsigned* __restrict__ emax, float* __restrict__ denom) {
    long long i = (long long)blockIdx.x * blockDim.x + threadIdx.x;
    if (i >= Etot * HEADS) return;
    const int head = (int)(i & 7);
    int s, d; edge_sd(ei, Eorig, i >> 3, &s, &d);
    const float v = leaky(e_src[s * HEADS + head] + e_dst[d * HEADS + head]);
    const float m = dec_f(emax[d * HEADS + head]);
    atomicAdd(&denom[d * HEADS + head], __expf(v - m));
}

// ---------- edge pass 3: alpha-weighted scatter of h[src] into outbuf[dst] ----------
__global__ void edge_scatter(const long long* __restrict__ ei, long long Eorig, long long Etot,
                             const float* __restrict__ e_src, const float* __restrict__ e_dst,
                             const unsigned* __restrict__ emax, const float* __restrict__ denom,
                             const float* __restrict__ h, float* __restrict__ outbuf) {
    long long i = (long long)blockIdx.x * blockDim.x + threadIdx.x;
    if (i >= Etot * HEADS) return;
    const int head = (int)(i & 7);
    int s, d; edge_sd(ei, Eorig, i >> 3, &s, &d);
    const float v = leaky(e_src[s * HEADS + head] + e_dst[d * HEADS + head]);
    const float m = dec_f(emax[d * HEADS + head]);
    const float alpha = __expf(v - m) / (denom[d * HEADS + head] + 1e-16f);
    const float4* hp = (const float4*)(h + (size_t)s * FDIM + head * CH);
    float* op = outbuf + (size_t)d * FDIM + head * CH;
#pragma unroll
    for (int q = 0; q < 4; ++q) {
        const float4 hv = hp[q];
        atomicAdd(op + q * 4 + 0, alpha * hv.x);
        atomicAdd(op + q * 4 + 1, alpha * hv.y);
        atomicAdd(op + q * 4 + 2, alpha * hv.z);
        atomicAdd(op + q * 4 + 3, alpha * hv.w);
    }
}

// ---------- epilogue: bias + ELU, and fused f16 conversion for next layer's WMMA ----------
__global__ void epilogue(float* __restrict__ outbuf, const float* __restrict__ bias,
                         _Float16* __restrict__ x16, int n_nodes) {
    int i = blockIdx.x * blockDim.x + threadIdx.x;
    if (i >= n_nodes * FDIM) return;
    float v = outbuf[i] + bias[i & (FDIM - 1)];
    v = v > 0.f ? v : (__expf(v) - 1.f);     // ELU
    outbuf[i] = v;
    x16[i] = (_Float16)v;
}

extern "C" void kernel_launch(void* const* d_in, const int* in_sizes, int n_in,
                              void* d_out, int out_size, void* d_ws, size_t ws_size,
                              hipStream_t stream) {
    const float*     x       = (const float*)d_in[0];   // [N,128]
    const float*     Ws      = (const float*)d_in[1];   // [3,128,128]
    const float*     att_src = (const float*)d_in[2];   // [3,8,16]
    const float*     att_dst = (const float*)d_in[3];   // [3,8,16]
    const float*     biases  = (const float*)d_in[4];   // [3,128]
    const float*     fc_w    = (const float*)d_in[5];   // [128,16]
    const float*     fc_b    = (const float*)d_in[6];   // [16]
    const long long* ei      = (const long long*)d_in[7]; // [2,E] int64

    const int       nN    = in_sizes[0] / FDIM;
    const long long Eorig = in_sizes[7] / 2;
    const long long Etot  = Eorig + nN;

    // workspace carve-out (256B aligned)
    char* ws = (char*)d_ws;
    size_t off = 0;
    auto alloc = [&](size_t bytes) -> void* {
        void* p = ws + off;
        off += (bytes + 255) & ~(size_t)255;
        return p;
    };
    float*     hbuf   = (float*)alloc((size_t)nN * FDIM * 4);   // GEMM output h
    float*     outbuf = (float*)alloc((size_t)nN * FDIM * 4);   // aggregation / next input
    _Float16*  x16    = (_Float16*)alloc((size_t)nN * FDIM * 2);
    _Float16*  w16    = (_Float16*)alloc((size_t)FDIM * FDIM * 2); // shuffled weights
    float*     e_src  = (float*)alloc((size_t)nN * HEADS * 4);
    float*     e_dst  = (float*)alloc((size_t)nN * HEADS * 4);
    unsigned*  emax   = (unsigned*)alloc((size_t)nN * HEADS * 4);
    float*     denom  = (float*)alloc((size_t)nN * HEADS * 4);

    const int T = 256;
    const int g_feat   = (nN * FDIM + T - 1) / T;
    const int g_score  = (nN * HEADS + T - 1) / T;
    const int g_edge   = (int)((Etot * HEADS + T - 1) / T);
    const int rowtiles = (nN + 15) / 16;
    const int g_gemm   = (rowtiles + 7) / 8;

    // layer-0 input -> f16
    cvt_f32_f16<<<g_feat, T, 0, stream>>>(x, x16, nN * FDIM);

    for (int l = 0; l < 3; ++l) {
        shuffle_w<<<(FDIM * FDIM + T - 1) / T, T, 0, stream>>>(Ws + (size_t)l * FDIM * FDIM,
                                                               w16, FDIM);
        init_layer<<<g_feat, T, 0, stream>>>(outbuf, emax, denom, nN);
        gemm128_wmma<<<g_gemm, T, 0, stream>>>(x16, w16, hbuf, nN);
        escore<<<g_score, T, 0, stream>>>(hbuf, att_src + l * HEADS * CH,
                                          att_dst + l * HEADS * CH, e_src, e_dst, nN);
        edge_max<<<g_edge, T, 0, stream>>>(ei, Eorig, Etot, e_src, e_dst, emax);
        edge_sum<<<g_edge, T, 0, stream>>>(ei, Eorig, Etot, e_src, e_dst, emax, denom);
        edge_scatter<<<g_edge, T, 0, stream>>>(ei, Eorig, Etot, e_src, e_dst, emax, denom,
                                               hbuf, outbuf);
        epilogue<<<g_feat, T, 0, stream>>>(outbuf, biases + l * FDIM, x16, nN);
    }

    // final FC: [N,128] @ [128,16] + b  (x16 already holds ELU'd layer-3 output)
    shuffle_w<<<(FDIM * CH + T - 1) / T, T, 0, stream>>>(fc_w, w16, CH);
    fc_wmma<<<g_gemm, T, 0, stream>>>(x16, w16, fc_b, (float*)d_out, nN);
}
